// HetGCN_71648644432155
// MI455X (gfx1250) — matrix operand order repaired and checked by reading, not compiled, not run on hardware
//
#include <hip/hip_runtime.h>
#include <hip/hip_bf16.h>

typedef __attribute__((ext_vector_type(16))) __bf16 v16bf;
typedef __attribute__((ext_vector_type(8)))  __bf16 v8bf;
typedef __attribute__((ext_vector_type(8)))  float  v8f;

static constexpr int N_U   = 100000;
static constexpr int N_I   = 100000;
static constexpr int EDGES = 500000;
static constexpr int H     = 128;

// ---------------------------------------------------------------------------
// utility kernels
// ---------------------------------------------------------------------------
__global__ void zero_f32(float* __restrict__ p, long n) {
    long i = (long)blockIdx.x * blockDim.x + threadIdx.x;
    long stride = (long)gridDim.x * blockDim.x;
    for (; i < n; i += stride) p[i] = 0.0f;
}

__global__ void deg_count(const int* __restrict__ src, const int* __restrict__ dst,
                          float* __restrict__ degs, float* __restrict__ degd, int E) {
    int i = blockIdx.x * blockDim.x + threadIdx.x;
    if (i >= E) return;
    unsafeAtomicAdd(&degs[src[i]], 1.0f);
    unsafeAtomicAdd(&degd[dst[i]], 1.0f);
}

__global__ void rsqrt_clip(float* __restrict__ d, int n) {
    int i = blockIdx.x * blockDim.x + threadIdx.x;
    if (i < n) d[i] = rsqrtf(fmaxf(d[i], 1.0f));
}

// Pack fp32 row-major W[K,128] into bf16 B-fragments laid out in a contiguous-k
// order matching load_afrag (lanes 0-15: K=kb..kb+15 at N=lane; lanes 16-31:
// K=kb+16..kb+31 at N=lane-16), so the GEMM reads 32B/lane fully coalesced.
__global__ void pack_weight(const float* __restrict__ W, __bf16* __restrict__ Wp,
                            int K, int Kpad) {
    int nkb = Kpad >> 5;
    int total = 8 * nkb * 512;               // 8 n-tiles * nkb k-blocks * 32 lanes * 16 elems
    int idx = blockIdx.x * blockDim.x + threadIdx.x;
    if (idx >= total) return;
    int e    = idx & 15;
    int lane = (idx >> 4) & 31;
    int blk  = idx >> 9;                     // t*nkb + kb
    int kb   = blk % nkb;
    int t    = blk / nkb;
    int k = (kb << 5) + (lane & 16) + e;
    int n = (t << 4) + (lane & 15);
    float v = (k < K) ? W[(size_t)k * 128 + n] : 0.0f;
    Wp[idx] = (__bf16)v;
}

// A-fragment fetch: contiguous-k order, lane<16 -> K kb..kb+15,
// lane>=16 -> +16. Two 16B LDS reads + shuffle.
__device__ __forceinline__ v16bf load_afrag(const __bf16* p) {
    const v8bf* ap = (const v8bf*)p;
    v8bf alo = ap[0];
    v8bf ahi = ap[1];
    return __builtin_shufflevector(alo, ahi,
               0,1,2,3,4,5,6,7,8,9,10,11,12,13,14,15);
}

// ---------------------------------------------------------------------------
// GEMM: Y[M,128] = (rownorm ? diag(rownorm) : I) * X[M,K] @ W[K,128] (+bias)
// bf16 WMMA, f32 accumulate.
// One block = 64 output rows (4 m-subtiles); 8 waves = 8 n-tiles.
// Each B fragment is reused across 4 accumulators; all 4 A fragments are
// fetched before the 4 WMMAs so LDS latency overlaps matrix issue.
// ---------------------------------------------------------------------------
__global__ void gemm_bf16_wmma(const float* __restrict__ X,
                               const __bf16* __restrict__ Wp,
                               const float* __restrict__ rownorm,
                               const float* __restrict__ bias,
                               float* __restrict__ Y,
                               int M, int K, int Kpad) {
    extern __shared__ __bf16 As[];           // 64 x Kpad bf16 (scaled A tile)
    const int tid = threadIdx.x;
    const int m0  = blockIdx.x << 6;

    // cooperative A-tile stage: fp32 -> (optional rsqrt(deg_out) scale) -> bf16
    const int tot = Kpad << 6;
    for (int idx = tid; idx < tot; idx += 256) {
        int r = idx / Kpad;
        int k = idx - r * Kpad;
        int row = m0 + r;
        float v = 0.0f;
        if (row < M && k < K) {
            v = X[(size_t)row * K + k];
            if (rownorm) v *= rownorm[row];
        }
        As[idx] = (__bf16)v;
    }
    __syncthreads();

    const int wave = tid >> 5;               // n-tile 0..7
    const int lane = tid & 31;
    const int half = lane >> 4;              // 0 or 1
    const int lr   = lane & 15;
    const int nkb  = Kpad >> 5;

    v8f c0 = {}, c1 = {}, c2 = {}, c3 = {};
    const __bf16* wbase = Wp + (((size_t)wave * nkb) << 9) + lane * 16;
    const __bf16* arow  = As + lr * Kpad + (half << 4);
    const int sstride = Kpad << 4;           // 16 rows of A tile

    for (int kb = 0; kb < nkb; ++kb) {
        // B frag: pre-swizzled, 32B contiguous per lane; reused 4x below
        v16bf b = *(const v16bf*)(wbase + ((size_t)kb << 9));
        const __bf16* ak = arow + (kb << 5);
        // batch all four A-fragment loads, then issue the four WMMAs
        v16bf a0 = load_afrag(ak);
        v16bf a1 = load_afrag(ak + sstride);
        v16bf a2 = load_afrag(ak + 2 * sstride);
        v16bf a3 = load_afrag(ak + 3 * sstride);
        c0 = __builtin_amdgcn_wmma_f32_16x16x32_bf16(false, a0, false, b, (short)0, c0, false, false);
        c1 = __builtin_amdgcn_wmma_f32_16x16x32_bf16(false, a1, false, b, (short)0, c1, false, false);
        c2 = __builtin_amdgcn_wmma_f32_16x16x32_bf16(false, a2, false, b, (short)0, c2, false, false);
        c3 = __builtin_amdgcn_wmma_f32_16x16x32_bf16(false, a3, false, b, (short)0, c3, false, false);
    }

    // D layout: VGPR r -> M = r + 8*half, N = lane%16 (per n-tile)
    const int n = (wave << 4) + lr;
    const float bv = bias ? bias[n] : 0.0f;
#pragma unroll
    for (int s = 0; s < 4; ++s) {
        v8f c = (s == 0) ? c0 : (s == 1) ? c1 : (s == 2) ? c2 : c3;
#pragma unroll
        for (int r = 0; r < 8; ++r) {
            int m = m0 + (s << 4) + r + (half << 3);
            if (m < M) Y[(size_t)m * 128 + n] = c[r] + bv;
        }
    }
}

// ---------------------------------------------------------------------------
// edge scatter: acc[dst] += h[src]  (128 lanes per edge, hw f32 atomics)
// ---------------------------------------------------------------------------
__global__ void scatter_add(const float* __restrict__ h, const int* __restrict__ src,
                            const int* __restrict__ dst, float* __restrict__ acc, int E) {
    int e = (blockIdx.x << 1) + (threadIdx.x >> 7);
    if (e >= E) return;
    int f = threadIdx.x & 127;
    int s = src[e], d = dst[e];
    unsafeAtomicAdd(&acc[((size_t)d << 7) + f], h[((size_t)s << 7) + f]);
}

// ---------------------------------------------------------------------------
// finalize: out = act(acc1*rs1 + b1) [+ act(acc2*rs2 + b2)]
// ---------------------------------------------------------------------------
__global__ void finalize2(const float* __restrict__ a1, const float* __restrict__ r1,
                          const float* __restrict__ b1,
                          const float* __restrict__ a2, const float* __restrict__ r2,
                          const float* __restrict__ b2,
                          float* __restrict__ out, int N, int relu) {
    long i = (long)blockIdx.x * 256 + threadIdx.x;
    long tot = (long)N * 128;
    if (i >= tot) return;
    int row = (int)(i >> 7), f = (int)(i & 127);
    float v1 = a1[i] * r1[row] + b1[f];
    float v2 = a2[i] * r2[row] + b2[f];
    if (relu) { v1 = fmaxf(v1, 0.0f); v2 = fmaxf(v2, 0.0f); }
    out[i] = v1 + v2;
}

__global__ void finalize1(const float* __restrict__ a1, const float* __restrict__ r1,
                          const float* __restrict__ b1,
                          float* __restrict__ out, int N, int relu) {
    long i = (long)blockIdx.x * 256 + threadIdx.x;
    long tot = (long)N * 128;
    if (i >= tot) return;
    int row = (int)(i >> 7), f = (int)(i & 127);
    float v = a1[i] * r1[row] + b1[f];
    if (relu) v = fmaxf(v, 0.0f);
    out[i] = v;
}

// ---------------------------------------------------------------------------
// host orchestration
// ---------------------------------------------------------------------------
extern "C" void kernel_launch(void* const* d_in, const int* in_sizes, int n_in,
                              void* d_out, int out_size, void* d_ws, size_t ws_size,
                              hipStream_t stream) {
    const float* feat_user = (const float*)d_in[0];
    const float* feat_item = (const float*)d_in[1];
    const int* src_uu = (const int*)d_in[2];
    const int* dst_uu = (const int*)d_in[3];
    const int* src_ui = (const int*)d_in[4];
    const int* dst_ui = (const int*)d_in[5];
    const int* src_iu = (const int*)d_in[6];
    const int* dst_iu = (const int*)d_in[7];
    const float* We_u = (const float*)d_in[8];
    const float* be_u = (const float*)d_in[9];
    const float* We_i = (const float*)d_in[10];
    const float* be_i = (const float*)d_in[11];
    // per-layer per-etype: d_in[12 + l*6 + 2*e] (e: 0=uu,1=ui,2=iu)
    const float* Wl[3][3]; const float* bl[3][3];
    for (int l = 0; l < 3; ++l)
        for (int e = 0; e < 3; ++e) {
            Wl[l][e] = (const float*)d_in[12 + l * 6 + 2 * e];
            bl[l][e] = (const float*)d_in[13 + l * 6 + 2 * e];
        }

    char* ws = (char*)d_ws;
    size_t off = 0;
    auto alloc = [&](size_t bytes) -> void* {
        void* p = ws + off;
        off = (off + bytes + 255) & ~(size_t)255;
        return p;
    };

    const size_t FEAT = (size_t)N_U * H * sizeof(float);   // 51.2 MB (N_U == N_I)
    float* hu    = (float*)alloc(FEAT);
    float* hi    = (float*)alloc(FEAT);
    float* htmp  = (float*)alloc(FEAT);
    float* accU1 = (float*)alloc(FEAT);
    float* accU2 = (float*)alloc(FEAT);
    float* accI  = (float*)alloc(FEAT);
    float* rs_s_uu = (float*)alloc(N_U * sizeof(float));
    float* rs_d_uu = (float*)alloc(N_U * sizeof(float));
    float* rs_s_ui = (float*)alloc(N_U * sizeof(float));
    float* rs_d_ui = (float*)alloc(N_I * sizeof(float));
    float* rs_s_iu = (float*)alloc(N_I * sizeof(float));
    float* rs_d_iu = (float*)alloc(N_U * sizeof(float));
    __bf16* WpEu = (__bf16*)alloc((size_t)8 * 8  * 512 * 2);   // K=256
    __bf16* WpEi = (__bf16*)alloc((size_t)8 * 10 * 512 * 2);   // K=300 -> Kpad 320
    __bf16* WpL[3][3];
    for (int l = 0; l < 3; ++l)
        for (int e = 0; e < 3; ++e)
            WpL[l][e] = (__bf16*)alloc((size_t)8 * 4 * 512 * 2);  // K=128

    const long NF = (long)N_U * H;

    // ---- degree / rsqrt normalization (graph structure only) ----
    zero_f32<<<256, 256, 0, stream>>>(rs_s_uu, N_U);
    zero_f32<<<256, 256, 0, stream>>>(rs_d_uu, N_U);
    zero_f32<<<256, 256, 0, stream>>>(rs_s_ui, N_U);
    zero_f32<<<256, 256, 0, stream>>>(rs_d_ui, N_I);
    zero_f32<<<256, 256, 0, stream>>>(rs_s_iu, N_I);
    zero_f32<<<256, 256, 0, stream>>>(rs_d_iu, N_U);
    int eb = (EDGES + 255) / 256;
    deg_count<<<eb, 256, 0, stream>>>(src_uu, dst_uu, rs_s_uu, rs_d_uu, EDGES);
    deg_count<<<eb, 256, 0, stream>>>(src_ui, dst_ui, rs_s_ui, rs_d_ui, EDGES);
    deg_count<<<eb, 256, 0, stream>>>(src_iu, dst_iu, rs_s_iu, rs_d_iu, EDGES);
    int nb = (N_U + 255) / 256;
    rsqrt_clip<<<nb, 256, 0, stream>>>(rs_s_uu, N_U);
    rsqrt_clip<<<nb, 256, 0, stream>>>(rs_d_uu, N_U);
    rsqrt_clip<<<nb, 256, 0, stream>>>(rs_s_ui, N_U);
    rsqrt_clip<<<nb, 256, 0, stream>>>(rs_d_ui, N_I);
    rsqrt_clip<<<nb, 256, 0, stream>>>(rs_s_iu, N_I);
    rsqrt_clip<<<nb, 256, 0, stream>>>(rs_d_iu, N_U);

    // ---- pack weights to bf16 WMMA B-fragment layout ----
    pack_weight<<<(8 * 8 * 512 + 255) / 256, 256, 0, stream>>>(We_u, WpEu, 256, 256);
    pack_weight<<<(8 * 10 * 512 + 255) / 256, 256, 0, stream>>>(We_i, WpEi, 300, 320);
    for (int l = 0; l < 3; ++l)
        for (int e = 0; e < 3; ++e)
            pack_weight<<<(8 * 4 * 512 + 255) / 256, 256, 0, stream>>>(Wl[l][e], WpL[l][e], 128, 128);

    // ---- HeteroLinear embed ----
    gemm_bf16_wmma<<<(N_U + 63) / 64, 256, 64 * 256 * 2, stream>>>(
        feat_user, WpEu, nullptr, be_u, hu, N_U, 256, 256);
    gemm_bf16_wmma<<<(N_I + 63) / 64, 256, 64 * 320 * 2, stream>>>(
        feat_item, WpEi, nullptr, be_i, hi, N_I, 300, 320);

    float* hu_out = (float*)d_out;
    float* hi_out = hu_out + (size_t)N_U * H;

    // ---- 3 GraphConv layers ----
    for (int l = 0; l < 3; ++l) {
        zero_f32<<<2048, 256, 0, stream>>>(accU1, NF);
        zero_f32<<<2048, 256, 0, stream>>>(accU2, NF);
        zero_f32<<<2048, 256, 0, stream>>>(accI,  NF);

        // uu: h = (hu * rs_out_uu) @ W -> scatter into accU1
        gemm_bf16_wmma<<<(N_U + 63) / 64, 256, 64 * 128 * 2, stream>>>(
            hu, WpL[l][0], rs_s_uu, nullptr, htmp, N_U, 128, 128);
        scatter_add<<<(EDGES + 1) / 2, 256, 0, stream>>>(htmp, src_uu, dst_uu, accU1, EDGES);

        // ui: from hu -> item accumulator
        gemm_bf16_wmma<<<(N_U + 63) / 64, 256, 64 * 128 * 2, stream>>>(
            hu, WpL[l][1], rs_s_ui, nullptr, htmp, N_U, 128, 128);
        scatter_add<<<(EDGES + 1) / 2, 256, 0, stream>>>(htmp, src_ui, dst_ui, accI, EDGES);

        // iu: from hi -> user accumulator 2
        gemm_bf16_wmma<<<(N_I + 63) / 64, 256, 64 * 128 * 2, stream>>>(
            hi, WpL[l][2], rs_s_iu, nullptr, htmp, N_I, 128, 128);
        scatter_add<<<(EDGES + 1) / 2, 256, 0, stream>>>(htmp, src_iu, dst_iu, accU2, EDGES);

        int relu = (l < 2) ? 1 : 0;
        float* ou = (l == 2) ? hu_out : hu;
        float* oi = (l == 2) ? hi_out : hi;
        finalize2<<<(int)((NF + 255) / 256), 256, 0, stream>>>(
            accU1, rs_d_uu, bl[l][0], accU2, rs_d_iu, bl[l][2], ou, N_U, relu);
        finalize1<<<(int)((NF + 255) / 256), 256, 0, stream>>>(
            accI, rs_d_ui, bl[l][1], oi, N_I, relu);
    }
}
